// Token_49323404427375
// MI455X (gfx1250) — matrix operand order repaired and checked by reading, not compiled, not run on hardware
//
#include <hip/hip_runtime.h>
#include <hip/hip_bf16.h>
#include <cstddef>

// ---------------------------------------------------------------------------
// Problem constants
// ---------------------------------------------------------------------------
#define Bc   32
#define Mc   16
#define Sc   128
#define Hc   768
#define Tc   34
#define Nc   (Bc * Mc)          // 512
#define NSc  (Nc * Sc)          // 65536
#define KTc  (Hc / 32)          // 24 k-steps of 32
#define JT_W (Hc / 16)          // 48 column tiles for HxH weights
#define JT_C 3                  // 48 padded cols for classifier (T=34 -> 48)

typedef __attribute__((ext_vector_type(16))) __bf16 v16bf;
typedef __attribute__((ext_vector_type(8)))  float  v8f;

// d_out layout (floats): [loss][logits_valid 65538*34][labels_valid 65538][ste 512*768]
static constexpr size_t OUT_LOGITS = 1;
static constexpr size_t OUT_LBL    = 1 + (size_t)Tc * (NSc + 2);        // 2228293
static constexpr size_t OUT_STE    = OUT_LBL + (size_t)(NSc + 2);       // 2293831

// workspace layout (bytes)
static constexpr size_t WSW_OFF  = 0;                                    // W_w bf16 swizzled
static constexpr size_t WSW_SZ   = (size_t)KTc * JT_W * 512 * 2;         // 1,179,648
static constexpr size_t CLSW_OFF = WSW_OFF + WSW_SZ;
static constexpr size_t CLSW_SZ  = (size_t)KTc * JT_C * 512 * 2;         // 73,728
static constexpr size_t LINW_OFF = CLSW_OFF + CLSW_SZ;
static constexpr size_t LINW_SZ  = WSW_SZ;
static constexpr size_t A_OFF    = LINW_OFF + LINW_SZ;                   // scores [NS]
static constexpr size_t A_SZ     = (size_t)NSc * 4;
static constexpr size_t TD_OFF   = A_OFF + A_SZ;                         // softmax [NS]
static constexpr size_t TD_SZ    = (size_t)NSc * 4;
static constexpr size_t HEAD_OFF = TD_OFF + TD_SZ;                       // [N,H]
static constexpr size_t HEAD_SZ  = (size_t)Nc * Hc * 4;
static constexpr size_t G_OFF    = HEAD_OFF + HEAD_SZ;                   // [N,H]
static constexpr size_t G_SZ     = (size_t)Nc * Hc * 4;
static constexpr size_t PRED_OFF = G_OFF + G_SZ;                         // int [NS]
static constexpr size_t PRED_SZ  = (size_t)NSc * 4;
static constexpr size_t ACC_OFF  = PRED_OFF + PRED_SZ;                   // 8 floats

// ---------------------------------------------------------------------------
// WMMA lane-layout helpers (CDNA5 16-bit 16x32 A / 32x16 B, wave32)
// ---------------------------------------------------------------------------
// A (MxK = 16x32): lanes 0-15 m=lane, e0..7 -> K0..7, e8..15 -> K16..23;
//                  lanes 16-31 hold K8..15 / K24..31.
__device__ __forceinline__ void a_slot(int m, int kk, int& lane, int& e) {
    lane = m + (((kk >> 3) & 1) << 4);
    e    = (kk & 7) + ((kk >= 16) ? 8 : 0);
}
// B (KxN = 32x16): n = lane&15; lanes 0-15 hold K0..7 / K16..23, lanes 16-31 K8..15 / K24..31 (same as A).
// (layout per ISA sparse-B table extrapolated to dense 32x16)

__device__ __forceinline__ v8f zero8() {
    v8f z;
#pragma unroll
    for (int i = 0; i < 8; ++i) z[i] = 0.f;
    return z;
}

// ---------------------------------------------------------------------------
// K_init: zero the scalar accumulators
// acc[0]=hinge_sum acc[1]=nll_sum acc[2]=valid_cnt acc[3]=E(label) acc[4]=E(pred)
// ---------------------------------------------------------------------------
__global__ void k_init(float* acc) {
    if (threadIdx.x < 8) acc[threadIdx.x] = 0.f;
}

// ---------------------------------------------------------------------------
// K_swizzle: f32 weight (optionally sum of two halves) -> bf16 WMMA-B layout.
// dst[((kt*JT + jt)*32 + lane)*16 + e]  <-  src[k*srcld + j]
// ---------------------------------------------------------------------------
__global__ void k_swizzle(const float* __restrict__ s0, const float* __restrict__ s1,
                          __bf16* __restrict__ dst, int srcld, int JT, int Jvalid) {
    int total = KTc * JT * 512;
    for (int t = blockIdx.x * blockDim.x + threadIdx.x; t < total;
         t += gridDim.x * blockDim.x) {
        int e = t & 15, lane = (t >> 4) & 31;
        int tile = t >> 9;
        int jt = tile % JT, kt = tile / JT;
        int n = lane & 15, khalf = lane >> 4;
        int k = kt * 32 + khalf * 8 + (e & 7) + ((e >> 3) << 4);
        int j = jt * 16 + n;
        float v = 0.f;
        if (j < Jvalid) {
            v = s0[(size_t)k * srcld + j];
            if (s1) v += s1[(size_t)k * srcld + j];
        }
        dst[t] = (__bf16)v;
    }
}

// ---------------------------------------------------------------------------
// K_attn: a[row] = v_b + sum_col v_w[col] * tanh( (x @ W_w)[row,col] + W_b[col] )
// 16-row tile per block, 8 waves x 6 col-tiles (covers all 768 cols). WMMA bf16.
// ---------------------------------------------------------------------------
__global__ __launch_bounds__(256) void k_attn(const float* __restrict__ x,
                                              const __bf16* __restrict__ Wsw,
                                              const float* __restrict__ Wb,
                                              const float* __restrict__ vw,
                                              const float* __restrict__ vb,
                                              float* __restrict__ a_out) {
    __shared__ __align__(32) __bf16 Albuf[32][16];
    __shared__ float red[16];
    const int tid  = threadIdx.x;
    const int wave = tid >> 5, lane = tid & 31;
    const int row0 = blockIdx.x * 16;

    v8f c[6];
#pragma unroll
    for (int jt = 0; jt < 6; ++jt) c[jt] = zero8();

    for (int kt = 0; kt < KTc; ++kt) {
        // stage A tile (16 rows x 32 k) as bf16 in WMMA-A lane layout
        int idx = tid * 2;
#pragma unroll
        for (int u = 0; u < 2; ++u) {
            int m = (idx + u) >> 5, kk = (idx + u) & 31;
            float v = x[(size_t)(row0 + m) * Hc + kt * 32 + kk];
            int dl, de; a_slot(m, kk, dl, de);
            Albuf[dl][de] = (__bf16)v;
        }
        __syncthreads();
        v16bf av = *reinterpret_cast<const v16bf*>(&Albuf[lane][0]);
#pragma unroll
        for (int jt = 0; jt < 6; ++jt) {
            int jtile = wave * 6 + jt;
            v16bf bv = *reinterpret_cast<const v16bf*>(
                Wsw + ((size_t)(kt * JT_W + jtile) * 32 + lane) * 16);
            c[jt] = __builtin_amdgcn_wmma_f32_16x16x32_bf16(
                false, av, false, bv, (short)0, c[jt], false, false);
        }
        __syncthreads();
    }

    // epilogue: tanh + dot with v_w, reduce across lanes/waves via LDS atomics
    float p[8];
#pragma unroll
    for (int r = 0; r < 8; ++r) p[r] = 0.f;
    const int nlo = lane & 15;
    const int rofs = (lane >= 16) ? 8 : 0;
#pragma unroll
    for (int jt = 0; jt < 6; ++jt) {
        int col = (wave * 6 + jt) * 16 + nlo;
        float wbv = Wb[col], vwv = vw[col];
#pragma unroll
        for (int r = 0; r < 8; ++r)
            p[r] += tanhf(c[jt][r] + wbv) * vwv;
    }
    if (tid < 16) red[tid] = 0.f;
    __syncthreads();
#pragma unroll
    for (int r = 0; r < 8; ++r) atomicAdd(&red[r + rofs], p[r]);
    __syncthreads();
    if (tid < 16) a_out[row0 + tid] = red[tid] + vb[0];
}

// ---------------------------------------------------------------------------
// K_softhead: per n, td = softmax_S(a), head = sum_s td[s] * x[n,s,:]
// ---------------------------------------------------------------------------
__global__ __launch_bounds__(128) void k_softhead(const float* __restrict__ a,
                                                  const float* __restrict__ x,
                                                  float* __restrict__ td,
                                                  float* __restrict__ head) {
    __shared__ float sb[Sc];
    const int n = blockIdx.x, s = threadIdx.x;
    float av = a[(size_t)n * Sc + s];
    sb[s] = av;
    __syncthreads();
    for (int off = 64; off > 0; off >>= 1) {
        if (s < off) sb[s] = fmaxf(sb[s], sb[s + off]);
        __syncthreads();
    }
    float mx = sb[0];
    __syncthreads();
    float e = __expf(av - mx);
    sb[s] = e;
    __syncthreads();
    for (int off = 64; off > 0; off >>= 1) {
        if (s < off) sb[s] += sb[s + off];
        __syncthreads();
    }
    float sum = sb[0];
    __syncthreads();
    float tdv = e / sum;
    td[(size_t)n * Sc + s] = tdv;
    sb[s] = tdv;
    __syncthreads();
#pragma unroll
    for (int ci = 0; ci < Hc / Sc; ++ci) {
        int h = s + ci * Sc;
        float acc = 0.f;
        for (int ss = 0; ss < Sc; ++ss)
            acc += sb[ss] * x[((size_t)n * Sc + ss) * Hc + h];
        head[(size_t)n * Hc + h] = acc;
    }
}

// ---------------------------------------------------------------------------
// K_gemm_g: g = head @ lin_comb  (512x768 @ 768x768), no bias (lin_b applied in xr)
// ---------------------------------------------------------------------------
__global__ __launch_bounds__(256) void k_gemm_g(const float* __restrict__ A,
                                                const __bf16* __restrict__ Bsw,
                                                float* __restrict__ g) {
    __shared__ __align__(32) __bf16 Albuf[32][16];
    const int tid  = threadIdx.x;
    const int wave = tid >> 5, lane = tid & 31;
    const int row0 = blockIdx.x * 16;

    v8f c[6];
#pragma unroll
    for (int jt = 0; jt < 6; ++jt) c[jt] = zero8();

    for (int kt = 0; kt < KTc; ++kt) {
        int idx = tid * 2;
#pragma unroll
        for (int u = 0; u < 2; ++u) {
            int m = (idx + u) >> 5, kk = (idx + u) & 31;
            float v = A[(size_t)(row0 + m) * Hc + kt * 32 + kk];
            int dl, de; a_slot(m, kk, dl, de);
            Albuf[dl][de] = (__bf16)v;
        }
        __syncthreads();
        v16bf av = *reinterpret_cast<const v16bf*>(&Albuf[lane][0]);
#pragma unroll
        for (int jt = 0; jt < 6; ++jt) {
            int jtile = wave * 6 + jt;
            v16bf bv = *reinterpret_cast<const v16bf*>(
                Bsw + ((size_t)(kt * JT_W + jtile) * 32 + lane) * 16);
            c[jt] = __builtin_amdgcn_wmma_f32_16x16x32_bf16(
                false, av, false, bv, (short)0, c[jt], false, false);
        }
        __syncthreads();
    }
    const int nlo = lane & 15;
    const int rofs = (lane >= 16) ? 8 : 0;
#pragma unroll
    for (int jt = 0; jt < 6; ++jt) {
        int col = (wave * 6 + jt) * 16 + nlo;
#pragma unroll
        for (int r = 0; r < 8; ++r)
            g[(size_t)(row0 + r + rofs) * Hc + col] = c[jt][r];
    }
}

// ---------------------------------------------------------------------------
// K_cls: xr = x + td*g + lin_b (rebuilt on the fly), logits = relu(xr@cls_w + cls_b).
// Also per-row hinge / NLL / argmax / label export. 4 independent waves / block.
// ---------------------------------------------------------------------------
__global__ __launch_bounds__(128) void k_cls(const float* __restrict__ x,
                                             const float* __restrict__ td,
                                             const float* __restrict__ g,
                                             const float* __restrict__ lin_b,
                                             const __bf16* __restrict__ Csw,
                                             const float* __restrict__ cls_b,
                                             const int* __restrict__ labels,
                                             const int* __restrict__ padp,
                                             float* __restrict__ out,
                                             int* __restrict__ pred_out,
                                             float* __restrict__ acc) {
    __shared__ __align__(32) __bf16 Abuf[4][32][16];
    __shared__ float Lbuf[4][16][48];
    const int tid  = threadIdx.x;
    const int wave = tid >> 5, lane = tid & 31;
    const int row0 = (blockIdx.x * 4 + wave) * 16;
    const int n    = row0 >> 7;  // 128 rows per n; constant within a 16-row tile

    v8f c[3];
#pragma unroll
    for (int jt = 0; jt < 3; ++jt) c[jt] = zero8();

    for (int kt = 0; kt < KTc; ++kt) {
        int col = kt * 32 + lane;
        float lb = lin_b[col];
        float gv = g[(size_t)n * Hc + col];
#pragma unroll
        for (int i = 0; i < 16; ++i) {
            int row = row0 + i;
            float xr = x[(size_t)row * Hc + col] + td[row] * gv + lb;
            int dl, de; a_slot(i, lane, dl, de);
            Abuf[wave][dl][de] = (__bf16)xr;
        }
        __syncthreads();
        v16bf av = *reinterpret_cast<const v16bf*>(&Abuf[wave][lane][0]);
#pragma unroll
        for (int jt = 0; jt < 3; ++jt) {
            v16bf bv = *reinterpret_cast<const v16bf*>(
                Csw + ((size_t)(kt * JT_C + jt) * 32 + lane) * 16);
            c[jt] = __builtin_amdgcn_wmma_f32_16x16x32_bf16(
                false, av, false, bv, (short)0, c[jt], false, false);
        }
        __syncthreads();
    }

    const int nlo = lane & 15;
    const int rofs = (lane >= 16) ? 8 : 0;
#pragma unroll
    for (int jt = 0; jt < 3; ++jt) {
        int j = jt * 16 + nlo;
        float cb = (j < Tc) ? cls_b[j] : 0.f;
#pragma unroll
        for (int r = 0; r < 8; ++r) {
            int row = r + rofs;
            float v = fmaxf(c[jt][r] + cb, 0.f);
            Lbuf[wave][row][j] = v;
            if (j < Tc)
                out[OUT_LOGITS + (size_t)(row0 + row + 1) * Tc + j] = v;
        }
    }
    __syncthreads();

    int pad = padp[0];
    if (lane < 16) {
        int grow = row0 + lane;
        int y = labels[grow];
        float sy = Lbuf[wave][lane][y];
        float mx = -1e30f, mxo = -1e30f;
        int arg = 0;
        for (int j = 0; j < Tc; ++j) {
            float v = Lbuf[wave][lane][j];
            if (v > mx) { mx = v; arg = j; }
            if (j != y && v > mxo) mxo = v;
        }
        float se = 0.f;
        for (int j = 0; j < Tc; ++j) se += __expf(Lbuf[wave][lane][j] - mx);
        float nll   = mx + __logf(se) - sy;
        float hinge = fmaxf(0.f, 1.f + mxo - sy);
        pred_out[grow] = arg;
        out[OUT_LBL + 1 + grow] = (float)y;
        if (y != pad) {
            atomicAdd(&acc[0], hinge);
            atomicAdd(&acc[1], nll);
            atomicAdd(&acc[2], 1.f);
        }
    }
}

// ---------------------------------------------------------------------------
// K_energy: local energy terms sum(xr . mat_local[label/pred]) + STE means.
// One block per n; xr recomputed on the fly.
// ---------------------------------------------------------------------------
__global__ __launch_bounds__(256) void k_energy(const float* __restrict__ x,
                                                const float* __restrict__ td,
                                                const float* __restrict__ g,
                                                const float* __restrict__ lin_b,
                                                const float* __restrict__ mat_local,
                                                const int* __restrict__ labels,
                                                const int* __restrict__ pred,
                                                const int* __restrict__ padp,
                                                float* __restrict__ out,
                                                float* __restrict__ acc) {
    __shared__ float r1[256], r2[256];
    const int n = blockIdx.x, t = threadIdx.x;
    const int pad = padp[0];

    float gv[3], lb[3], steac[3];
#pragma unroll
    for (int ci = 0; ci < 3; ++ci) {
        int col = t + ci * 256;
        gv[ci] = g[(size_t)n * Hc + col];
        lb[ci] = lin_b[col];
        steac[ci] = 0.f;
    }
    float al = 0.f, ap = 0.f;
    int cnt = 0;
    for (int s = 0; s < Sc; ++s) {
        int row = n * Sc + s;
        float tdv = td[row];
        int l = labels[row], p = pred[row];
        int trig = (p != pad);
        cnt += trig;
#pragma unroll
        for (int ci = 0; ci < 3; ++ci) {
            int col = t + ci * 256;
            float xr = x[(size_t)row * Hc + col] + tdv * gv[ci] + lb[ci];
            al += xr * mat_local[(size_t)l * Hc + col];
            ap += xr * mat_local[(size_t)p * Hc + col];
            if (trig) steac[ci] += xr;
        }
    }
    float cf = (float)cnt;
#pragma unroll
    for (int ci = 0; ci < 3; ++ci) {
        int col = t + ci * 256;
        out[OUT_STE + (size_t)n * Hc + col] = (cnt > 0) ? steac[ci] / cf : 0.f;
    }
    r1[t] = al; r2[t] = ap;
    __syncthreads();
    for (int off = 128; off > 0; off >>= 1) {
        if (t < off) { r1[t] += r1[t + off]; r2[t] += r2[t + off]; }
        __syncthreads();
    }
    if (t == 0) {
        atomicAdd(&acc[3], r1[0]);
        atomicAdd(&acc[4], r2[0]);
    }
}

// ---------------------------------------------------------------------------
// K_labenergy: cross-batch term sum_{n,m} mat_label[y[n,S-2], y[m,S-1]] via
// class-count histograms (for labels and predictions).
// ---------------------------------------------------------------------------
__global__ __launch_bounds__(256) void k_labenergy(const int* __restrict__ labels,
                                                   const int* __restrict__ pred,
                                                   const float* __restrict__ mat_label,
                                                   float* __restrict__ acc) {
    __shared__ int cA[Tc], cB[Tc], pA[Tc], pB[Tc];
    __shared__ float rb[256];
    const int t = threadIdx.x;
    if (t < Tc) { cA[t] = 0; cB[t] = 0; pA[t] = 0; pB[t] = 0; }
    __syncthreads();
    for (int nn = t; nn < Nc; nn += 256) {
        atomicAdd(&cA[labels[nn * Sc + (Sc - 2)]], 1);
        atomicAdd(&cB[labels[nn * Sc + (Sc - 1)]], 1);
        atomicAdd(&pA[pred[nn * Sc + (Sc - 2)]], 1);
        atomicAdd(&pB[pred[nn * Sc + (Sc - 1)]], 1);
    }
    __syncthreads();
    float al = 0.f, ap = 0.f;
    for (int idx = t; idx < Tc * Tc; idx += 256) {
        int u = idx % Tc, v = idx / Tc;
        float m = mat_label[v * Tc + u];
        al += (float)(cA[v] * cB[u]) * m;
        ap += (float)(pA[v] * pB[u]) * m;
    }
    rb[t] = al;
    __syncthreads();
    for (int off = 128; off > 0; off >>= 1) {
        if (t < off) rb[t] += rb[t + off];
        __syncthreads();
    }
    if (t == 0) atomicAdd(&acc[3], rb[0]);
    __syncthreads();
    rb[t] = ap;
    __syncthreads();
    for (int off = 128; off > 0; off >>= 1) {
        if (t < off) rb[t] += rb[t + off];
        __syncthreads();
    }
    if (t == 0) atomicAdd(&acc[4], rb[0]);
}

// ---------------------------------------------------------------------------
// K_final: combine losses; duplicate first/last logits rows + label endpoints.
// ---------------------------------------------------------------------------
__global__ void k_final(const float* __restrict__ acc,
                        const int* __restrict__ labels,
                        float* __restrict__ out) {
    const int t = threadIdx.x;
    if (t < Tc) {
        out[OUT_LOGITS + t] = out[OUT_LOGITS + Tc + t];                       // dup row 0
        out[OUT_LOGITS + (size_t)(NSc + 1) * Tc + t] =
            out[OUT_LOGITS + (size_t)NSc * Tc + t];                           // dup last
    }
    if (t == 40) {
        out[OUT_LBL] = (float)labels[0];
        out[OUT_LBL + NSc + 1] = (float)labels[NSc - 1];
    }
    if (t == 63) {
        float nv  = fmaxf(acc[2], 1.f);
        float lh  = acc[0] / nv;
        float lce = acc[1] / nv;
        float lte = fmaxf(0.f, lh + acc[3] - acc[4]);
        out[0] = lte + 0.5f * lce;   // ENERGY_WEIGHT=1, RATIO_LOSS_TOKEN_PLUS=0.5
    }
}

// ---------------------------------------------------------------------------
// Launch
// ---------------------------------------------------------------------------
extern "C" void kernel_launch(void* const* d_in, const int* in_sizes, int n_in,
                              void* d_out, int out_size, void* d_ws, size_t ws_size,
                              hipStream_t stream) {
    const float* x        = (const float*)d_in[0];
    const int*   labels   = (const int*)d_in[1];
    const int*   padp     = (const int*)d_in[4];
    const float* W_w      = (const float*)d_in[6];
    const float* W_b      = (const float*)d_in[7];
    const float* v_w      = (const float*)d_in[8];
    const float* v_b      = (const float*)d_in[9];
    const float* cls_w    = (const float*)d_in[10];
    const float* cls_b    = (const float*)d_in[11];
    const float* matloc   = (const float*)d_in[12];
    const float* matlab   = (const float*)d_in[13];
    const float* lin_w    = (const float*)d_in[14];
    const float* lin_b    = (const float*)d_in[15];
    float* out = (float*)d_out;
    char*  ws  = (char*)d_ws;

    __bf16* wsw   = (__bf16*)(ws + WSW_OFF);
    __bf16* clssw = (__bf16*)(ws + CLSW_OFF);
    __bf16* linsw = (__bf16*)(ws + LINW_OFF);
    float*  abuf  = (float*)(ws + A_OFF);
    float*  tdb   = (float*)(ws + TD_OFF);
    float*  head  = (float*)(ws + HEAD_OFF);
    float*  gbuf  = (float*)(ws + G_OFF);
    int*    pred  = (int*)(ws + PRED_OFF);
    float*  acc   = (float*)(ws + ACC_OFF);

    k_init<<<1, 64, 0, stream>>>(acc);

    // weight preprocessing: bf16 + WMMA-B swizzle
    {
        int totW = KTc * JT_W * 512;
        k_swizzle<<<(totW + 255) / 256, 256, 0, stream>>>(W_w, nullptr, wsw, Hc, JT_W, Hc);
        // lin_comb = lin_w[:H,:] + lin_w[H:,:]  (since ctx = [head, head])
        k_swizzle<<<(totW + 255) / 256, 256, 0, stream>>>(lin_w, lin_w + (size_t)Hc * Hc,
                                                          linsw, Hc, JT_W, Hc);
        int totC = KTc * JT_C * 512;
        k_swizzle<<<(totC + 255) / 256, 256, 0, stream>>>(cls_w, nullptr, clssw, Tc, JT_C, Tc);
    }

    // attention scores a = v . tanh(x @ W + b) + vb
    k_attn<<<NSc / 16, 256, 0, stream>>>(x, wsw, W_b, v_w, v_b, abuf);

    // softmax over S + head = sum td * x
    k_softhead<<<Nc, Sc, 0, stream>>>(abuf, x, tdb, head);

    // g = head @ lin_comb
    k_gemm_g<<<Nc / 16, 256, 0, stream>>>(head, linsw, gbuf);

    // logits + per-token losses/argmax
    k_cls<<<NSc / 64, 128, 0, stream>>>(x, tdb, gbuf, lin_b, clssw, cls_b,
                                        labels, padp, out, pred, acc);

    // local energies + sentence trigger embeddings
    k_energy<<<Nc, 256, 0, stream>>>(x, tdb, gbuf, lin_b, matloc, labels, pred,
                                     padp, out, acc);

    // cross-batch label energy term
    k_labenergy<<<1, 256, 0, stream>>>(labels, pred, matlab, acc);

    // final combine + boundary duplication
    k_final<<<1, 64, 0, stream>>>(acc, labels, out);
}
